// RNN_34342558498949
// MI455X (gfx1250) — matrix operand order repaired and checked by reading, not compiled, not run on hardware
//
#include <hip/hip_runtime.h>

typedef __attribute__((ext_vector_type(16))) _Float16 v16h;
typedef __attribute__((ext_vector_type(2)))  _Float16 h2v;
typedef __attribute__((ext_vector_type(8)))  float    v8f;

#define RNN_B 512
#define RNN_T 2000
#define RNN_D 3
#define RNN_H 50
// Augmented K layout per row of h_aug[16][64] (f16, row-major in LDS):
//   k = 0..49   : h_{t-1}
//   k = 50,51   : 0
//   k = 52..54  : x_t           (8-byte aligned slot: x+one is a single b64 store)
//   k = 55      : 1.0 (bias row)
//   k = 56..63  : 0

__device__ __forceinline__ float waug_rec(int k, int n,
                                          const float* __restrict__ Whh,
                                          const float* __restrict__ Wih,
                                          const float* __restrict__ bih,
                                          const float* __restrict__ bhh) {
  if (n >= RNN_H) return 0.0f;
  if (k < RNN_H) return Whh[n * RNN_H + k];                   // h @ W_hh^T
  if (k >= 52 && k <= 54) return Wih[n * RNN_D + (k - 52)];   // x @ W_ih^T
  if (k == 55) return bih[n] + bhh[n];                        // bias via 1-slot
  return 0.0f;
}

struct f3 { float a, b, c; };

__global__ __launch_bounds__(32)
void rnn_scan_wmma(const float* __restrict__ x,
                   const float* __restrict__ Wih,
                   const float* __restrict__ Whh,
                   const float* __restrict__ bih,
                   const float* __restrict__ bhh,
                   const float* __restrict__ Wout,
                   const float* __restrict__ bout,
                   float* __restrict__ y) {
  __shared__ _Float16 hbuf[16 * 64];   // one 16-row batch tile, row-major [m][k]

  const int lane  = threadIdx.x;       // wave32
  const int m     = lane & 15;         // batch row within tile (B-operand N idx)
  const int hi    = lane >> 4;         // lane half selects K sub-range
  const int brow  = blockIdx.x * 16;

  // ---- Loop-invariant A-operand fragments (W_aug^T) in contiguous K order (k = 16*hi + i), same as B.
  v16h Arec[4][2];                     // [n'-tile][k-chunk]  recurrence weights
  v16h Aout[2];                        // W_out rows (d = 0..2), zero elsewhere
  #pragma unroll
  for (int mt = 0; mt < 4; ++mt) {
    #pragma unroll
    for (int kc = 0; kc < 2; ++kc) {
      v16h f;
      #pragma unroll
      for (int i = 0; i < 16; ++i) {
        int v = i >> 1, hb = i & 1;
        int k = kc * 32 + hi * 16 + 2 * v + hb;
        int n = mt * 16 + m;           // output feature index n'
        f[i] = (_Float16)waug_rec(k, n, Whh, Wih, bih, bhh);
      }
      Arec[mt][kc] = f;
    }
  }
  #pragma unroll
  for (int kc = 0; kc < 2; ++kc) {
    v16h f;
    #pragma unroll
    for (int i = 0; i < 16; ++i) {
      int v = i >> 1, hb = i & 1;
      int k = kc * 32 + hi * 16 + 2 * v + hb;
      int d = m;
      f[i] = (_Float16)((d < RNN_D && k < RNN_H) ? Wout[d * RNN_H + k] : 0.0f);
    }
    Aout[kc] = f;
  }
  v8f cb;                              // C-init for output proj = b_out
  #pragma unroll
  for (int v = 0; v < 8; ++v)
    cb[v] = (hi == 0 && v < RNN_D) ? bout[v] : 0.0f;

  // ---- init LDS: h_{-1}=0, stage x_0 + constant 1 ----
  for (int i = lane; i < (16 * 64) / 8; i += 32)
    ((uint4*)hbuf)[i] = make_uint4(0u, 0u, 0u, 0u);
  __syncthreads();
  if (hi == 0) {
    const float* xp = x + ((size_t)(brow + m) * RNN_T + 0) * RNN_D;
    union { _Float16 h[4]; uint2 u; } q;
    q.h[0] = (_Float16)xp[0]; q.h[1] = (_Float16)xp[1];
    q.h[2] = (_Float16)xp[2]; q.h[3] = (_Float16)1.0f;
    *(uint2*)(hbuf + m * 64 + 52) = q.u;
  }

  const v8f zeroC = {};
  const h2v z2 = {};                   // packed f16 zero
  for (int t = 0; t <= RNN_T; ++t) {
    __syncthreads();
    // B operand (h_aug^T, 64x16): lane m reads its own contiguous 32B K-slice.
    v16h bk0 = *(const v16h*)(hbuf + m * 64 + hi * 16);        // K = 0..31
    v16h bk1 = *(const v16h*)(hbuf + m * 64 + 32 + hi * 16);   // K = 32..63

    // Output projection of h_{t-1}: reuses the same B fragments (zero A rows
    // kill the x/bias K-slots):  y^T = W_out_aug^T x h_aug^T + b_out
    v8f cy = __builtin_amdgcn_wmma_f32_16x16x32_f16(false, Aout[0], false, bk0,
                                                    (short)0, cb, false, false);
    cy     = __builtin_amdgcn_wmma_f32_16x16x32_f16(false, Aout[1], false, bk1,
                                                    (short)0, cy, false, false);
    if (t > 0 && hi == 0) {
      f3 o; o.a = cy[0]; o.b = cy[1]; o.c = cy[2];   // vgpr v = output dim d
      *(f3*)(y + ((size_t)(brow + m) * RNN_T + (t - 1)) * RNN_D) = o;
    }
    if (t == RNN_T) break;

    // Recurrence: h_t^T = relu(W_aug^T x h_aug^T).
    // Independent accumulators for the 4 n'-tiles: 8 WMMAs issue back-to-back
    // and fill each other's D->A/B hazard slots.
    v8f acc[4];
    #pragma unroll
    for (int mt = 0; mt < 4; ++mt)
      acc[mt] = __builtin_amdgcn_wmma_f32_16x16x32_f16(false, Arec[mt][0], false, bk0,
                                                       (short)0, zeroC, false, false);
    #pragma unroll
    for (int mt = 0; mt < 4; ++mt)
      acc[mt] = __builtin_amdgcn_wmma_f32_16x16x32_f16(false, Arec[mt][1], false, bk1,
                                                       (short)0, acc[mt], false, false);
    #pragma unroll
    for (int mt = 0; mt < 4; ++mt) {
      // Pack to f16 first (v_cvt_pk_f16_f32, RNE), then relu in packed f16:
      // cvt output is canonical, so elementwise max is a single v_pk_max_f16.
      union { h2v h2[4]; uint4 u; } p;
      #pragma unroll
      for (int i = 0; i < 4; ++i) {
        h2v v;
        v.x = (_Float16)acc[mt][2 * i];
        v.y = (_Float16)acc[mt][2 * i + 1];
        p.h2[i] = __builtin_elementwise_max(v, z2);
      }
      // C/D layout: lane holds batch row m, 8 consecutive n' -> one b128 store
      *(uint4*)(hbuf + m * 64 + mt * 16 + hi * 8) = p.u;
    }
    // Stage x_{t+1} + bias one (overwrites the zeros just stored at k=52..55).
    if (hi == 0 && t + 1 < RNN_T) {
      const float* xp = x + ((size_t)(brow + m) * RNN_T + (t + 1)) * RNN_D;
      union { _Float16 h[4]; uint2 u; } q;
      q.h[0] = (_Float16)xp[0]; q.h[1] = (_Float16)xp[1];
      q.h[2] = (_Float16)xp[2]; q.h[3] = (_Float16)1.0f;
      *(uint2*)(hbuf + m * 64 + 52) = q.u;
    }
  }
}

extern "C" void kernel_launch(void* const* d_in, const int* in_sizes, int n_in,
                              void* d_out, int out_size, void* d_ws, size_t ws_size,
                              hipStream_t stream) {
  const float* x    = (const float*)d_in[0];
  const float* Wih  = (const float*)d_in[1];
  const float* Whh  = (const float*)d_in[2];
  const float* bih  = (const float*)d_in[3];
  const float* bhh  = (const float*)d_in[4];
  const float* Wout = (const float*)d_in[5];
  const float* bout = (const float*)d_in[6];
  float* yout = (float*)d_out;

  dim3 grid(RNN_B / 16);   // 32 independent batch tiles
  dim3 block(32);          // one wave32 each
  rnn_scan_wmma<<<grid, block, 0, stream>>>(x, Wih, Whh, bih, bhh, Wout, bout, yout);
}
